// WindowAttention_89515708383938
// MI455X (gfx1250) — compile-verified, matrix operand order checked by reading, not run on hardware
//
#include <hip/hip_runtime.h>
#include <math.h>

typedef _Float16 half_t;
typedef _Float16 v16h __attribute__((ext_vector_type(16)));
typedef _Float16 v8h  __attribute__((ext_vector_type(8)));
typedef float    v8f  __attribute__((ext_vector_type(8)));

// LDS row strides (halves) padded to kill ds_load_b128 bank conflicts:
// 40 halves = 80B, 72 halves = 144B -> per-lane 16B accesses land on distinct
// 16B granules mod 256B; both are multiples of 16B so vector loads stay aligned.
#define LDA32 40
#define LDA64 72

// Single-instruction XOR butterfly via ds_swizzle_b32 (group-of-32 mode):
// src_lane = ((lane & 0x1f) | 0) ^ XORMASK ; offset = (XORMASK<<10) | 0x1f.
template <int OFF>
__device__ inline float swz_xor(float x) {
    return __int_as_float(__builtin_amdgcn_ds_swizzle(__float_as_int(x), OFF));
}
#define SWZ1  0x041f  // xor 1
#define SWZ2  0x081f  // xor 2
#define SWZ4  0x101f  // xor 4
#define SWZ8  0x201f  // xor 8

// ---------------------------------------------------------------------------
// Load a 16x32 (MxK) WMMA A/B operand (16-bit) from LDS.
// ISA 7.12.2: lanes 0-15 hold K = {0..7, 16..23}, lanes 16-31 hold K = {8..15, 24..31}
// for matrix row/col = lane & 15.  stride/kbase in halves.
// ---------------------------------------------------------------------------
__device__ inline v16h lds_op16(const half_t* base, int stride, int kbase) {
    const int lane = threadIdx.x & 31;
    const half_t* p = base + (lane & 15) * stride + kbase + ((lane >> 4) << 3);
    v8h a = *(const v8h*)p;          // K chunk 0
    v8h b = *(const v8h*)(p + 16);   // K chunk 1 (+16 in K)
    return __builtin_shufflevector(a, b, 0,1,2,3,4,5,6,7,8,9,10,11,12,13,14,15);
}

// Stage 16 contiguous K-elements into LDS as f16 (converting if needed).
__device__ inline void stage16(half_t* dst, const float* src) {
    const float4* s = (const float4*)src;
    float4 x0 = s[0], x1 = s[1], x2 = s[2], x3 = s[3];
    v8h h0 = { (half_t)x0.x,(half_t)x0.y,(half_t)x0.z,(half_t)x0.w,
               (half_t)x1.x,(half_t)x1.y,(half_t)x1.z,(half_t)x1.w };
    v8h h1 = { (half_t)x2.x,(half_t)x2.y,(half_t)x2.z,(half_t)x2.w,
               (half_t)x3.x,(half_t)x3.y,(half_t)x3.z,(half_t)x3.w };
    *(v8h*)dst = h0;
    *((v8h*)dst + 1) = h1;
}
__device__ inline void stage16(half_t* dst, const half_t* src) {
    const v8h* s = (const v8h*)src;
    v8h a = s[0], b = s[1];
    *(v8h*)dst = a;
    *((v8h*)dst + 1) = b;
}

__global__ void f32_to_f16_kernel(const float* __restrict__ src,
                                  half_t* __restrict__ dst, int n) {
    int i = blockIdx.x * blockDim.x + threadIdx.x;
    int stride = gridDim.x * blockDim.x;
    for (; i < n; i += stride) dst[i] = (half_t)src[i];
}

// ---------------------------------------------------------------------------
// C[M,N] = A[M,K] * Bt[N,K]^T + bias[N]
// Block: 128x128 tile, 256 threads = 8 waves as 4(M) x 2(N); wave -> 32x64.
// M, N, K multiples of 128/128/32 (true here: M=100352, N in {1536,512}, K=512).
// ---------------------------------------------------------------------------
template <typename AT, typename OT>
__global__ __launch_bounds__(256) void gemm_wmma_kernel(
    const AT* __restrict__ A, const half_t* __restrict__ Bt,
    const float* __restrict__ bias, OT* __restrict__ C, int K, int N) {
    __shared__ half_t As[128 * LDA32];
    __shared__ half_t Bs[128 * LDA32];

    const int tid  = threadIdx.x;
    const int lane = tid & 31;
    const int w    = tid >> 5;
    const int wm   = w >> 1;   // 0..3
    const int wn   = w & 1;    // 0..1
    const long m0  = (long)blockIdx.x * 128;
    const long n0  = (long)blockIdx.y * 128;

    v8f zero = {};
    v8f acc[2][4];
#pragma unroll
    for (int a = 0; a < 2; ++a)
#pragma unroll
        for (int b = 0; b < 4; ++b) acc[a][b] = zero;

    const int srow = tid >> 1;        // 0..127
    const int skb  = (tid & 1) * 16;  // 0 or 16

    for (int k0 = 0; k0 < K; k0 += 32) {
        __syncthreads();
        stage16(&As[srow * LDA32 + skb], A  + (m0 + srow) * (long)K + k0 + skb);
        stage16(&Bs[srow * LDA32 + skb], Bt + (n0 + srow) * (long)K + k0 + skb);
        __builtin_prefetch(A  + (m0 + srow) * (long)K + k0 + 32 + skb, 0, 1);
        __builtin_prefetch(Bt + (n0 + srow) * (long)K + k0 + 32 + skb, 0, 1);
        __syncthreads();

        v16h aop[2];
#pragma unroll
        for (int mt = 0; mt < 2; ++mt)
            aop[mt] = lds_op16(&As[(wm * 32 + mt * 16) * LDA32], LDA32, 0);
#pragma unroll
        for (int nt = 0; nt < 4; ++nt) {
            v16h bop = lds_op16(&Bs[(wn * 64 + nt * 16) * LDA32], LDA32, 0);
#pragma unroll
            for (int mt = 0; mt < 2; ++mt)
                acc[mt][nt] = __builtin_amdgcn_wmma_f32_16x16x32_f16(
                    false, aop[mt], false, bop, (short)0, acc[mt][nt], false, false);
        }
    }

    // C/D layout: lane&15 -> N col, reg r -> M row (+8 for lanes 16-31)
    const int rbase = (lane >> 4) << 3;
#pragma unroll
    for (int mt = 0; mt < 2; ++mt) {
#pragma unroll
        for (int nt = 0; nt < 4; ++nt) {
            const long col = n0 + wn * 64 + nt * 16 + (lane & 15);
            const float bv = bias[col];
#pragma unroll
            for (int r = 0; r < 8; ++r) {
                const long row = m0 + wm * 32 + mt * 16 + rbase + r;
                C[row * (long)N + col] = (OT)(acc[mt][nt][r] + bv);
            }
        }
    }
}

// ---------------------------------------------------------------------------
// Fused window attention: one block = one (batch-window, head).
// 49 tokens padded to 64; head_dim = 32.  4 waves, wave w owns score rows
// [16w, 16w+16).  S = QK^T*scale + Add; softmax; O = P V.
// Add[i][j] = relpos_bias(i,j,h) + mask(b%64,i,j) pre-staged in LDS.
// ---------------------------------------------------------------------------
__global__ __launch_bounds__(128) void win_attn_kernel(
    const half_t* __restrict__ qkv,        // (2048*49, 1536) = q|k|v per row
    const float* __restrict__ bias_table,  // (13*13, 16)
    const float* __restrict__ mask,        // (64, 49, 49)
    half_t* __restrict__ ctx) {            // (2048*49, 512)
    __shared__ half_t Qs[64 * LDA32];
    __shared__ half_t Ks[64 * LDA32];
    __shared__ half_t Vt[32 * LDA64];  // transposed: Vt[d][n]
    __shared__ half_t Ps[64 * LDA64];
    __shared__ float  Add[64 * 65];

    const int bh   = blockIdx.x;
    const int b    = bh >> 4;
    const int h    = bh & 15;
    const int tid  = threadIdx.x;
    const int lane = tid & 31;
    const int wv   = tid >> 5;

    // ---- Stage Q,K vectorized (v8h chunks); zero-pad rows 49..63. ----
    for (int idx = tid; idx < 64 * 4; idx += 128) {
        const int n = idx >> 2, c = (idx & 3) * 8;
        v8h zq = {}, zk = {};
        if (n < 49) {
            const half_t* p = qkv + ((long)b * 49 + n) * 1536 + h * 32 + c;
            zq = *(const v8h*)p;
            zk = *(const v8h*)(p + 512);
        }
        *(v8h*)&Qs[n * LDA32 + c] = zq;
        *(v8h*)&Ks[n * LDA32 + c] = zk;
    }
    // ---- Stage V transposed (scalar; write Vt[d][n]). ----
    for (int idx = tid; idx < 64 * 32; idx += 128) {
        const int n = idx >> 5, d = idx & 31;
        half_t v = (half_t)0.f;
        if (n < 49)
            v = qkv[((long)b * 49 + n) * 1536 + 1024 + h * 32 + d];
        Vt[d * LDA64 + n] = v;
    }
    // ---- Stage combined additive term (bias gather + mask, coalesced). ----
    {
        const float* mrow = mask + (b & 63) * 49 * 49;
        for (int idx = tid; idx < 64 * 64; idx += 128) {
            const int i = idx >> 6, j = idx & 63;
            float a;
            if (i < 49 && j < 49) {
                const int ih = i / 7, iw = i % 7, jh = j / 7, jw = j % 7;
                const int bidx = (ih - jh + 6) * 13 + (iw - jw + 6);
                a = bias_table[bidx * 16 + h] + mrow[i * 49 + j];
            } else {
                a = (i < 49) ? -1e30f : 0.f;  // pad cols -> -inf; pad rows benign
            }
            Add[i * 65 + j] = a;
        }
    }
    __syncthreads();

    const int mrow0 = wv * 16;
    const int rbase = (lane >> 4) << 3;
    const int jlow  = lane & 15;

    // S = Q @ K^T  (64x64 padded, K-dim = 32, one WMMA per 16x16 tile)
    v16h aop = lds_op16(&Qs[mrow0 * LDA32], LDA32, 0);
    v8f  zero = {};
    float sv[4][8];
#pragma unroll
    for (int ct = 0; ct < 4; ++ct) {
        v16h bop = lds_op16(&Ks[ct * 16 * LDA32], LDA32, 0);
        v8f s = __builtin_amdgcn_wmma_f32_16x16x32_f16(false, aop, false, bop,
                                                       (short)0, zero, false, false);
#pragma unroll
        for (int r = 0; r < 8; ++r) sv[ct][r] = s[r];
    }

    const float inv_scale = 0.17677669529663687f;  // 1/sqrt(32)
#pragma unroll
    for (int ct = 0; ct < 4; ++ct) {
        const int j = ct * 16 + jlow;
#pragma unroll
        for (int r = 0; r < 8; ++r) {
            const int i = mrow0 + rbase + r;
            sv[ct][r] = sv[ct][r] * inv_scale + Add[i * 65 + j];
        }
    }

    // Row softmax across 16-lane halves: single-instruction XOR butterflies
    // via ds_swizzle_b32 (masks 1,2,4,8 stay within each half).
#pragma unroll
    for (int r = 0; r < 8; ++r) {
        float m = fmaxf(fmaxf(sv[0][r], sv[1][r]), fmaxf(sv[2][r], sv[3][r]));
        m = fmaxf(m, swz_xor<SWZ1>(m));
        m = fmaxf(m, swz_xor<SWZ2>(m));
        m = fmaxf(m, swz_xor<SWZ4>(m));
        m = fmaxf(m, swz_xor<SWZ8>(m));
        float sum = 0.f;
#pragma unroll
        for (int ct = 0; ct < 4; ++ct) { sv[ct][r] = __expf(sv[ct][r] - m); sum += sv[ct][r]; }
        sum += swz_xor<SWZ1>(sum);
        sum += swz_xor<SWZ2>(sum);
        sum += swz_xor<SWZ4>(sum);
        sum += swz_xor<SWZ8>(sum);
        const float rs = 1.f / sum;
#pragma unroll
        for (int ct = 0; ct < 4; ++ct) sv[ct][r] *= rs;
    }

    // P -> LDS (f16).  Rows [mrow0, mrow0+16) written & re-read by this wave only.
#pragma unroll
    for (int ct = 0; ct < 4; ++ct) {
        const int j = ct * 16 + jlow;
#pragma unroll
        for (int r = 0; r < 8; ++r) {
            const int i = mrow0 + rbase + r;
            Ps[i * LDA64 + j] = (half_t)sv[ct][r];
        }
    }

    // O = P @ V : 16x32 out per wave, K = 64 (two 32-steps).
    v8f acc2[2] = {zero, zero};
#pragma unroll
    for (int ks = 0; ks < 2; ++ks) {
        v16h pop = lds_op16(&Ps[mrow0 * LDA64], LDA64, ks * 32);
#pragma unroll
        for (int nt = 0; nt < 2; ++nt) {
            v16h vop = lds_op16(&Vt[nt * 16 * LDA64], LDA64, ks * 32);
            acc2[nt] = __builtin_amdgcn_wmma_f32_16x16x32_f16(false, pop, false, vop,
                                                              (short)0, acc2[nt], false, false);
        }
    }

#pragma unroll
    for (int nt = 0; nt < 2; ++nt) {
        const int d = nt * 16 + jlow;
#pragma unroll
        for (int r = 0; r < 8; ++r) {
            const int i = mrow0 + rbase + r;
            if (i < 49)
                ctx[((long)b * 49 + i) * 512 + h * 32 + d] = (half_t)acc2[nt][r];
        }
    }
}

// ---------------------------------------------------------------------------
extern "C" void kernel_launch(void* const* d_in, const int* in_sizes, int n_in,
                              void* d_out, int out_size, void* d_ws, size_t ws_size,
                              hipStream_t stream) {
    (void)in_sizes; (void)n_in; (void)out_size; (void)ws_size;
    const float* x          = (const float*)d_in[0];  // (2048,49,512)
    const float* mask       = (const float*)d_in[1];  // (64,49,49)
    const float* qkv_w      = (const float*)d_in[2];  // (1536,512)
    const float* qkv_b      = (const float*)d_in[3];  // (1536,)
    const float* proj_w     = (const float*)d_in[4];  // (512,512)
    const float* proj_b     = (const float*)d_in[5];  // (512,)
    const float* bias_table = (const float*)d_in[6];  // (13,13,16)
    float* out = (float*)d_out;                       // (2048,49,512)

    char* ws = (char*)d_ws;
    half_t* wqkv_h  = (half_t*)(ws);                        // 1536*512*2   = 1.5 MB
    half_t* wproj_h = (half_t*)(ws + 1572864);              // 512*512*2    = 0.5 MB
    half_t* qkv_h   = (half_t*)(ws + 2097152);              // 100352*1536*2 = 294 MB
    half_t* ctx_h   = (half_t*)(ws + 2097152 + 308281344);  // 100352*512*2  = 98 MB

    f32_to_f16_kernel<<<768, 256, 0, stream>>>(qkv_w,  wqkv_h,  1536 * 512);
    f32_to_f16_kernel<<<256, 256, 0, stream>>>(proj_w, wproj_h, 512 * 512);

    // QKV projection: (100352,512) x (512,1536) -> f16
    gemm_wmma_kernel<float, half_t><<<dim3(784, 12), 256, 0, stream>>>(
        x, wqkv_h, qkv_b, qkv_h, 512, 1536);

    // Fused attention per (window, head)
    win_attn_kernel<<<2048 * 16, 128, 0, stream>>>(qkv_h, bias_table, mask, ctx_h);

    // Output projection: (100352,512) x (512,512) -> f32 out
    gemm_wmma_kernel<half_t, float><<<dim3(784, 4), 256, 0, stream>>>(
        ctx_h, wproj_h, proj_b, out, 512, 512);
}
